// LSTM_Decoder_80453327389020
// MI455X (gfx1250) — matrix operand order repaired and checked by reading, not compiled, not run on hardware
//
#include <hip/hip_runtime.h>

// ---------------------------------------------------------------------------
// LSTM decoder (layers independent; only layer 2 reaches the output, f-gate
// dead because c0=0). Split-bf16 (hi+lo) WMMA GEMMs, f32 accumulate.
// ---------------------------------------------------------------------------

typedef __attribute__((ext_vector_type(16))) __bf16 v16bf;
typedef __attribute__((ext_vector_type(8)))  float  v8f;

union Frag {
    v16bf v;
    uint4 q[2];
};

#define HD     1024       // hidden / input size
#define BATCH  64
#define TSTEPS 256
#define GROWS  3072       // packed gate rows: i(0..1023) g(1024..2047) o(2048..3071)

// ---- bf16 helpers ---------------------------------------------------------
__device__ __forceinline__ unsigned short f2bf_rn(float x) {
    unsigned int u = __float_as_uint(x);
    u += 0x7FFFu + ((u >> 16) & 1u);           // round-to-nearest-even
    return (unsigned short)(u >> 16);
}
__device__ __forceinline__ float bf2f(unsigned short h) {
    return __uint_as_float(((unsigned int)h) << 16);
}
__device__ __forceinline__ void split_bf16(float x, unsigned short& hi, unsigned short& lo) {
    hi = f2bf_rn(x);
    lo = f2bf_rn(x - bf2f(hi));
}

__device__ __forceinline__ float fsigmoid(float x) { return 1.0f / (1.0f + __expf(-x)); }
__device__ __forceinline__ float ftanh(float x) {
    float e = __expf(-2.0f * x);
    return (1.0f - e) / (1.0f + e);
}

// ---- WMMA fragment loads (contiguous k order, same for A 16x32 and B 32x16) --
// A: lane row = M = lane&15; lanes 0-15 hold K=0..15, lanes 16-31
//    hold K=16..31  -> two 16B chunks at k0+2*khalf and k0+2*khalf+8.
__device__ __forceinline__ void loadA(Frag& f, const unsigned short* row, int k0, int khalf) {
    f.q[0] = *(const uint4*)(row + k0 + 2 * khalf);
    f.q[1] = *(const uint4*)(row + k0 + 2 * khalf + 8);
}
// B: lane column = N = lane&15; lanes 0-15 hold K=0..15, lanes 16-31 K=16..31
//    -> one contiguous 32B run at k0 + (lane>>4)*16 of the row-major weight row.
__device__ __forceinline__ void loadB(Frag& f, const unsigned short* row, int kb) {
    f.q[0] = *(const uint4*)(row + kb);
    f.q[1] = *(const uint4*)(row + kb + 8);
}

__device__ __forceinline__ v8f wmma_bf16(const Frag& a, const Frag& b, v8f c) {
    return __builtin_amdgcn_wmma_f32_16x16x32_bf16(false, a.v, false, b.v,
                                                   (short)0, c, false, false);
}

// ---------------------------------------------------------------------------
// Preprocessing kernels (run once per call; weights -> bf16 hi/lo, biases, h0)
// ---------------------------------------------------------------------------
__global__ void prep_wg(const float* __restrict__ W_ih,
                        unsigned short* __restrict__ hiA,
                        unsigned short* __restrict__ loA) {
    int idx = blockIdx.x * blockDim.x + threadIdx.x;     // < GROWS*HD
    if (idx >= GROWS * HD) return;
    int p = idx >> 10;                                   // packed gate row
    int k = idx & (HD - 1);
    int src = (p < 1024) ? p : p + 1024;                 // skip f-gate rows
    float x = W_ih[8388608 + src * HD + k];              // layer 2 base
    unsigned short h, l; split_bf16(x, h, l);
    hiA[idx] = h; loA[idx] = l;
}

__global__ void prep_wo(const float* __restrict__ W_out,
                        unsigned short* __restrict__ hiA,
                        unsigned short* __restrict__ loA) {
    int idx = blockIdx.x * blockDim.x + threadIdx.x;     // < HD*HD
    if (idx >= HD * HD) return;
    unsigned short h, l; split_bf16(W_out[idx], h, l);
    hiA[idx] = h; loA[idx] = l;
}

__global__ void prep_bias(const float* __restrict__ b_ih, const float* __restrict__ b_hh,
                          const float* __restrict__ b_out,
                          float* __restrict__ biasg, float* __restrict__ boutf) {
    int idx = blockIdx.x * blockDim.x + threadIdx.x;
    if (idx < GROWS) {
        int src = (idx < 1024) ? idx : idx + 1024;
        biasg[idx] = b_ih[8192 + src] + b_hh[8192 + src];   // layer 2 base
    } else if (idx < GROWS + HD) {
        boutf[idx - GROWS] = b_out[idx - GROWS];
    }
}

__global__ void prep_h(const float* __restrict__ dh,
                       float* __restrict__ out_tail,
                       unsigned short* __restrict__ h_hi,
                       unsigned short* __restrict__ h_lo) {
    int idx = blockIdx.x * blockDim.x + threadIdx.x;     // < 3*BATCH*HD
    if (idx >= 3 * BATCH * HD) return;
    float x = dh[idx];
    out_tail[idx] = x;                                   // returned decoder_hidden
    if (idx >= 2 * BATCH * HD) {                         // layer 2 -> h0
        unsigned short h, l; split_bf16(x, h, l);
        h_hi[idx - 2 * BATCH * HD] = h;
        h_lo[idx - 2 * BATCH * HD] = l;
    }
}

// ---------------------------------------------------------------------------
// Fused step: blocks [0,rec_blocks) compute h_{t+1} = cell(h_t)  (i/g/o GEMM)
//             blocks [rec_blocks, ...) compute out_{t-1} = h_t @ W_out^T + b
// Each wave32 owns one 16x16 C tile; K loop of 32 with split-bf16 products.
// ---------------------------------------------------------------------------
__global__ __launch_bounds__(128, 1) void lstm_step(
    const unsigned short* __restrict__ h_hi, const unsigned short* __restrict__ h_lo,
    unsigned short* __restrict__ hn_hi, unsigned short* __restrict__ hn_lo,
    const unsigned short* __restrict__ Wg_hi, const unsigned short* __restrict__ Wg_lo,
    const unsigned short* __restrict__ Wo_hi, const unsigned short* __restrict__ Wo_lo,
    const float* __restrict__ biasg, const float* __restrict__ boutf,
    float* __restrict__ out_t, int rec_blocks)
{
    const int lane  = threadIdx.x & 31;
    const int wave  = threadIdx.x >> 5;
    const int r15   = lane & 15;
    const int khalf = (lane >> 4) << 3;   // A chunk select (0 or 8)
    const int kq    = (lane >> 4) << 4;   // B K-half select (0 or 16)

    if ((int)blockIdx.x < rec_blocks) {
        // ---------------- recurrence: gates i/g/o ----------------
        int idx   = blockIdx.x * 4 + wave;     // 0..255
        int mtile = idx >> 6;                  // 0..3
        int jtile = idx & 63;                  // 0..63

        const unsigned short* aHi = h_hi + (mtile * 16 + r15) * HD;
        const unsigned short* aLo = h_lo + (mtile * 16 + r15) * HD;
        const unsigned short* bIh = Wg_hi + (jtile * 16 + r15) * HD;
        const unsigned short* bIl = Wg_lo + (jtile * 16 + r15) * HD;
        const unsigned short* bGh = bIh + 1024 * HD;
        const unsigned short* bGl = bIl + 1024 * HD;
        const unsigned short* bOh = bIh + 2048 * HD;
        const unsigned short* bOl = bIl + 2048 * HD;

        v8f accI = {0.f,0.f,0.f,0.f,0.f,0.f,0.f,0.f};
        v8f accG = accI, accO = accI;

        for (int k0 = 0; k0 < HD; k0 += 32) {
            Frag ah, al, fih, fil, fgh, fgl, foh, fol;
            loadA(ah, aHi, k0, khalf);
            loadA(al, aLo, k0, khalf);
            int kb = k0 + kq;
            loadB(fih, bIh, kb); loadB(fil, bIl, kb);
            loadB(fgh, bGh, kb); loadB(fgl, bGl, kb);
            loadB(foh, bOh, kb); loadB(fol, bOl, kb);
            // split product: hi*hi + hi*lo + lo*hi (lo*lo ~2^-18, dropped)
            accI = wmma_bf16(ah, fih, accI);
            accI = wmma_bf16(ah, fil, accI);
            accI = wmma_bf16(al, fih, accI);
            accG = wmma_bf16(ah, fgh, accG);
            accG = wmma_bf16(ah, fgl, accG);
            accG = wmma_bf16(al, fgh, accG);
            accO = wmma_bf16(ah, foh, accO);
            accO = wmma_bf16(ah, fol, accO);
            accO = wmma_bf16(al, foh, accO);
        }

        int j  = jtile * 16 + r15;
        float bI = biasg[j], bG = biasg[1024 + j], bO = biasg[2048 + j];
        int mbase = mtile * 16 + ((lane >> 4) << 3);   // C layout: lanes16-31 -> M+8
        if (hn_hi) {
            #pragma unroll
            for (int r = 0; r < 8; r++) {
                float vi = accI[r] + bI;
                float vg = accG[r] + bG;
                float vo = accO[r] + bO;
                float c  = fsigmoid(vi) * ftanh(vg);   // f*c0 == 0
                float hv = fsigmoid(vo) * ftanh(c);
                unsigned short hh, hl; split_bf16(hv, hh, hl);
                int off = (mbase + r) * HD + j;
                hn_hi[off] = hh; hn_lo[off] = hl;
            }
        }
    } else {
        // ---------------- output projection ----------------
        if (!out_t) return;
        int idx   = ((int)blockIdx.x - rec_blocks) * 4 + wave;
        int mtile = idx >> 6;
        int ntile = idx & 63;

        const unsigned short* aHi = h_hi + (mtile * 16 + r15) * HD;
        const unsigned short* aLo = h_lo + (mtile * 16 + r15) * HD;
        const unsigned short* bH  = Wo_hi + (ntile * 16 + r15) * HD;
        const unsigned short* bL  = Wo_lo + (ntile * 16 + r15) * HD;

        v8f acc = {0.f,0.f,0.f,0.f,0.f,0.f,0.f,0.f};
        for (int k0 = 0; k0 < HD; k0 += 32) {
            Frag ah, al, bh, bl;
            loadA(ah, aHi, k0, khalf);
            loadA(al, aLo, k0, khalf);
            int kb = k0 + kq;
            loadB(bh, bH, kb);
            loadB(bl, bL, kb);
            acc = wmma_bf16(ah, bh, acc);
            acc = wmma_bf16(ah, bl, acc);
            acc = wmma_bf16(al, bh, acc);
        }
        int n  = ntile * 16 + r15;
        float bn = boutf[n];
        int mbase = mtile * 16 + ((lane >> 4) << 3);
        #pragma unroll
        for (int r = 0; r < 8; r++)
            out_t[(mbase + r) * HD + n] = acc[r] + bn;
    }
}

// ---------------------------------------------------------------------------
extern "C" void kernel_launch(void* const* d_in, const int* in_sizes, int n_in,
                              void* d_out, int out_size, void* d_ws, size_t ws_size,
                              hipStream_t stream) {
    const float* dh    = (const float*)d_in[0];   // [3,64,1024]
    const float* W_ih  = (const float*)d_in[1];   // [3,4096,1024]
    const float* b_ih  = (const float*)d_in[2];   // [3,4096]
    const float* b_hh  = (const float*)d_in[3];   // [3,4096]
    const float* W_out = (const float*)d_in[4];   // [1024,1024]
    const float* b_out = (const float*)d_in[5];   // [1024]
    float* out = (float*)d_out;                   // [256*64*1024] + [3*64*1024]

    char* ws = (char*)d_ws;
    size_t off = 0;
    auto wsalloc = [&](size_t bytes) -> void* {
        void* p = ws + off;
        off = (off + bytes + 255) & ~(size_t)255;
        return p;
    };
    unsigned short* Wg_hi = (unsigned short*)wsalloc((size_t)GROWS * HD * 2);
    unsigned short* Wg_lo = (unsigned short*)wsalloc((size_t)GROWS * HD * 2);
    unsigned short* Wo_hi = (unsigned short*)wsalloc((size_t)HD * HD * 2);
    unsigned short* Wo_lo = (unsigned short*)wsalloc((size_t)HD * HD * 2);
    float* biasg = (float*)wsalloc(GROWS * 4);
    float* boutf = (float*)wsalloc(HD * 4);
    unsigned short* hHi[2], *hLo[2];
    hHi[0] = (unsigned short*)wsalloc(BATCH * HD * 2);
    hLo[0] = (unsigned short*)wsalloc(BATCH * HD * 2);
    hHi[1] = (unsigned short*)wsalloc(BATCH * HD * 2);
    hLo[1] = (unsigned short*)wsalloc(BATCH * HD * 2);

    // --- preprocessing (weights -> bf16 hi/lo, bias sums, h0, hidden copy) ---
    prep_wg  <<<(GROWS * HD + 255) / 256, 256, 0, stream>>>(W_ih, Wg_hi, Wg_lo);
    prep_wo  <<<(HD * HD + 255) / 256,    256, 0, stream>>>(W_out, Wo_hi, Wo_lo);
    prep_bias<<<(GROWS + HD + 255) / 256, 256, 0, stream>>>(b_ih, b_hh, b_out, biasg, boutf);
    prep_h   <<<(3 * BATCH * HD + 255) / 256, 256, 0, stream>>>(
        dh, out + (size_t)TSTEPS * BATCH * HD, hHi[0], hLo[0]);

    // --- step 0: recurrence only (h0 -> h1) ---
    lstm_step<<<64, 128, 0, stream>>>(hHi[0], hLo[0], hHi[1], hLo[1],
                                      Wg_hi, Wg_lo, Wo_hi, Wo_lo, biasg, boutf,
                                      nullptr, 64);
    // --- steps 1..255: fused (h_t -> h_{t+1}) + (out_{t-1} = h_t @ W_out^T) ---
    for (int t = 1; t < TSTEPS; t++) {
        int ci = t & 1, ni = ci ^ 1;
        lstm_step<<<128, 128, 0, stream>>>(hHi[ci], hLo[ci], hHi[ni], hLo[ni],
                                           Wg_hi, Wg_lo, Wo_hi, Wo_lo, biasg, boutf,
                                           out + (size_t)(t - 1) * BATCH * HD, 64);
    }
    // --- final: out_255 from h_256 (in buffer 0 since 256 is even) ---
    lstm_step<<<64, 128, 0, stream>>>(hHi[0], hLo[0], nullptr, nullptr,
                                      Wg_hi, Wg_lo, Wo_hi, Wo_lo, biasg, boutf,
                                      out + (size_t)(TSTEPS - 1) * BATCH * HD, 0);
}
